// CausalSelfAttention_15006615733272
// MI455X (gfx1250) — matrix operand order repaired and checked by reading, not compiled, not run on hardware
//
#include <hip/hip_runtime.h>

#define D_MODEL 1024
#define NHEAD   16
#define DK      64
#define BATCH   4
#define SEQ     1024

typedef __attribute__((ext_vector_type(16))) __bf16 v16bf;
typedef __attribute__((ext_vector_type(8)))  float  v8f;

union BF16Frag { unsigned short us[16]; uint4 q[2]; v16bf v; };

#if defined(__gfx1250__) && __has_builtin(__builtin_amdgcn_global_load_async_to_lds_b128)
#define HAVE_ASYNC_LDS 1
#endif

#ifdef HAVE_ASYNC_LDS
typedef int v4i_vs __attribute__((vector_size(16)));
typedef __attribute__((address_space(1))) v4i_vs* g_v4i_p;   // global int4*
typedef __attribute__((address_space(3))) v4i_vs* l_v4i_p;   // LDS int4*
#endif

// 16B global->LDS copy: async (ASYNCcnt) when available, else direct.
__device__ __forceinline__ void copy16_g2l(const void* gsrc, void* lds_dst) {
#ifdef HAVE_ASYNC_LDS
  __builtin_amdgcn_global_load_async_to_lds_b128(
      (g_v4i_p)gsrc, (l_v4i_p)lds_dst, 0, 0);
#else
  *(uint4*)lds_dst = *(const uint4*)gsrc;
#endif
}
__device__ __forceinline__ void async_wait0() {
#ifdef HAVE_ASYNC_LDS
#if __has_builtin(__builtin_amdgcn_s_wait_asynccnt)
  __builtin_amdgcn_s_wait_asynccnt(0);
#else
  asm volatile("s_wait_asynccnt 0x0" ::: "memory");
#endif
#endif
}

__device__ __forceinline__ unsigned short f2bf(float f) {
  unsigned u = __float_as_uint(f);
  unsigned r = u + 0x7FFFu + ((u >> 16) & 1u);   // round-to-nearest-even
  return (unsigned short)(r >> 16);
}

__device__ __forceinline__ uint2 f2bf4(float4 f) {
  uint2 r;
  r.x = (unsigned)f2bf(f.x) | ((unsigned)f2bf(f.y) << 16);
  r.y = (unsigned)f2bf(f.z) | ((unsigned)f2bf(f.w) << 16);
  return r;
}

// A-matrix fragment (16-bit, 16xK row-major in LDS): per ISA layout,
// us[0..7] = K 8*half .. +7, us[8..15] = K 16+8*half .. +7
__device__ __forceinline__ v16bf frag_A(const unsigned short* rowbase, int half) {
  BF16Frag f;
  f.q[0] = *(const uint4*)(rowbase + 8 * half);
  f.q[1] = *(const uint4*)(rowbase + 16 + 8 * half);
  return f.v;
}

// B-matrix fragment (16-bit, KxN, stored TRANSPOSED in LDS as [n][k]):
// us[0..7] = K 8*half .. +7, us[8..15] = K 16+8*half .. +7 -> two 16B loads
__device__ __forceinline__ v16bf frag_B(const unsigned short* colbase, int half) {
  BF16Frag f;
  f.q[0] = *(const uint4*)(colbase + 8 * half);
  f.q[1] = *(const uint4*)(colbase + 16 + 8 * half);
  return f.v;
}

__device__ __forceinline__ void atomicMaxF_lds(float* addr, float val) {
  unsigned* ua = (unsigned*)addr;
  unsigned old = *ua;
  while (__uint_as_float(old) < val) {
    unsigned assumed = old;
    old = atomicCAS(ua, assumed, __float_as_uint(val));
    if (old == assumed) break;
  }
}

// ---------------------------------------------------------------------------
// Kernel 1: fused QKV projections.  Y = x @ W^T + b.
//   Q,K -> [B,H,S,DK] bf16;  V -> [B,H,DK,S] bf16 (d-major for attn B-tiles).
// 64x64 output tile per 8-wave WG; K-step 64 -> 4 WMMAs per wave per barrier.
// ---------------------------------------------------------------------------
__global__ __launch_bounds__(256) void k_qkv(
    const float* __restrict__ x,
    const float* __restrict__ Wq, const float* __restrict__ bq,
    const float* __restrict__ Wk, const float* __restrict__ bk,
    const float* __restrict__ Wv, const float* __restrict__ bv,
    unsigned short* __restrict__ Qb, unsigned short* __restrict__ Kb,
    unsigned short* __restrict__ Vb)
{
  const int c0  = blockIdx.x * 64;
  const int r0  = blockIdx.y * 64;
  const int mat = blockIdx.z;
  const float* W    = (mat == 0) ? Wq : (mat == 1) ? Wk : Wv;
  const float* bias = (mat == 0) ? bq : (mat == 1) ? bk : bv;
  unsigned short* dst = (mat == 0) ? Qb : (mat == 1) ? Kb : Vb;

  __shared__ alignas(16) unsigned short As[64][64];  // x tile, row-major
  __shared__ alignas(16) unsigned short Bs[64][64];  // Bs[c][k] = W[c0+c][kc+k]

  const int tid  = threadIdx.x;
  const int wave = tid >> 5, lane = tid & 31;
  const int half = lane >> 4, ln = lane & 15;
  const int row_t = wave >> 1;          // 0..3
  const int colt0 = (wave & 1) * 2;     // 0 or 2

  v8f acc0 = {}; v8f acc1 = {};

  for (int kc = 0; kc < D_MODEL; kc += 64) {
    // 64x64 floats = 1024 float4 each; 4 per thread per tile
#pragma unroll
    for (int t = 0; t < 4; ++t) {
      int i = tid + t * 256;
      int r = i >> 4, c4 = (i & 15) * 4;
      float4 fa = *(const float4*)&x[(size_t)(r0 + r) * D_MODEL + kc + c4];
      *(uint2*)&As[r][c4] = f2bf4(fa);
      float4 fb = *(const float4*)&W[(size_t)(c0 + r) * D_MODEL + kc + c4];
      *(uint2*)&Bs[r][c4] = f2bf4(fb);
    }
    __syncthreads();

    v16bf a0  = frag_A(&As[row_t * 16 + ln][0], half);
    v16bf a1  = frag_A(&As[row_t * 16 + ln][32], half);
    v16bf b00 = frag_B(&Bs[colt0 * 16 + ln][0], half);
    v16bf b01 = frag_B(&Bs[colt0 * 16 + ln][32], half);
    v16bf b10 = frag_B(&Bs[(colt0 + 1) * 16 + ln][0], half);
    v16bf b11 = frag_B(&Bs[(colt0 + 1) * 16 + ln][32], half);
    acc0 = __builtin_amdgcn_wmma_f32_16x16x32_bf16(false, a0, false, b00,
                                                   (short)0, acc0, false, false);
    acc0 = __builtin_amdgcn_wmma_f32_16x16x32_bf16(false, a1, false, b01,
                                                   (short)0, acc0, false, false);
    acc1 = __builtin_amdgcn_wmma_f32_16x16x32_bf16(false, a0, false, b10,
                                                   (short)0, acc1, false, false);
    acc1 = __builtin_amdgcn_wmma_f32_16x16x32_bf16(false, a1, false, b11,
                                                   (short)0, acc1, false, false);
    __syncthreads();
  }

#pragma unroll
  for (int i = 0; i < 8; ++i) {
    int row  = r0 + row_t * 16 + i + 8 * half;   // C/D layout: M = i + 8*half
    int bidx = row >> 10, s = row & 1023;
#pragma unroll
    for (int j = 0; j < 2; ++j) {
      int col = c0 + (colt0 + j) * 16 + ln;
      float y = (j ? acc1[i] : acc0[i]) + bias[col];
      int hh = col >> 6, d = col & 63;
      size_t idx = (mat < 2)
        ? (((size_t)(bidx * NHEAD + hh)) * SEQ + s) * DK + d     // [B,H,S,DK]
        : (((size_t)(bidx * NHEAD + hh)) * DK + d) * SEQ + s;    // [B,H,DK,S]
      dst[idx] = f2bf(y);
    }
  }
}

// ---------------------------------------------------------------------------
// Kernel 2: causal attention for one (b, h, 32-query-row tile).
// Double-buffered async (ASYNCcnt) K/V tile pipeline, 1 barrier/iteration:
//   iter i:  wait(own async)=0 ; barrier ; issue tile i+1 into other buffer
//            (that buffer was last read before this barrier) ; compute tile i
// ---------------------------------------------------------------------------
__global__ __launch_bounds__(256) void k_attn(
    const unsigned short* __restrict__ Qb,
    const unsigned short* __restrict__ Kb,
    const unsigned short* __restrict__ Vb,   // [B,H,DK,S]
    float* __restrict__ attn, float* __restrict__ ctx)
{
  const int q0 = blockIdx.x * 32;
  const int h  = blockIdx.y, b = blockIdx.z;
  const int bh = b * NHEAD + h;

  __shared__ alignas(16) unsigned short Qs[32][64];      // [q][dk]
  __shared__ alignas(16) unsigned short Ks[2][64][64];   // [buf][key][dk]
  __shared__ alignas(16) unsigned short Es[2][32][32];   // [buf][q][key]
  __shared__ alignas(16) unsigned short Vs[2][64][32];   // [buf][dk][key]
  __shared__ float rmax[32];
  __shared__ float rsum[32];

  const int tid  = threadIdx.x;
  const int wave = tid >> 5, lane = tid & 31;
  const int half = lane >> 4, ln = lane & 15;
  const int mi = wave >> 2;   // 0..1 : query sub-tile
  const int ni = wave & 3;    // 0..3 : key / dk sub-tile

  const int kend2 = q0 + 32;               // causal bound (multiple of 32)
  const int kend1 = (kend2 + 63) & ~63;    // pass-1 granularity, <= SEQ
  const int nit1  = kend1 >> 6;
  const int nit2  = kend2 >> 5;

  if (tid < 32) { rmax[tid] = -1e30f; rsum[tid] = 0.f; }

  {  // Qs: 2048 shorts = 256 x 16B, 1 per thread
    int r = tid >> 3, c8 = (tid & 7) * 8;
    copy16_g2l(&Qb[((size_t)bh * SEQ + q0 + r) * DK + c8], &Qs[r][c8]);
  }
  {  // K tile 0: 4096 shorts = 512 x 16B, 2 per thread
#pragma unroll
    for (int t = 0; t < 2; ++t) {
      int i = tid + t * 256;
      int r = i >> 3, c8 = (i & 7) * 8;
      copy16_g2l(&Kb[((size_t)bh * SEQ + r) * DK + c8], &Ks[0][r][c8]);
    }
  }
  async_wait0();
  __syncthreads();

  v16bf aq0 = frag_A(&Qs[mi * 16 + ln][0], half);        // dk 0..31
  v16bf aq1 = frag_A(&Qs[mi * 16 + ln][32], half);       // dk 32..63

  // ---- pass 1: raw masked scores + row max ----
  for (int it = 0; it < nit1; ++it) {
    const int kc = it << 6, cur = it & 1;
    if (it) { async_wait0(); __syncthreads(); }
    if (it + 1 < nit1) {   // issue next tile; its buffer was read before barrier
      const int nk = (it + 1) << 6, nb = cur ^ 1;
#pragma unroll
      for (int t = 0; t < 2; ++t) {
        int i = tid + t * 256;
        int r = i >> 3, c8 = (i & 7) * 8;
        copy16_g2l(&Kb[((size_t)bh * SEQ + nk + r) * DK + c8], &Ks[nb][r][c8]);
      }
    }
    // B[kk][n] = K[kc + ni*16 + n][kk]; Ks is [key][dk] -> contiguous in kk
    v16bf b0 = frag_B(&Ks[cur][ni * 16 + ln][0], half);
    v16bf b1 = frag_B(&Ks[cur][ni * 16 + ln][32], half);
    v8f sc = {};
    sc = __builtin_amdgcn_wmma_f32_16x16x32_bf16(false, aq0, false, b0,
                                                 (short)0, sc, false, false);
    sc = __builtin_amdgcn_wmma_f32_16x16x32_bf16(false, aq1, false, b1,
                                                 (short)0, sc, false, false);
#pragma unroll
    for (int i = 0; i < 8; ++i) {
      int rl = mi * 16 + i + 8 * half;
      int q  = q0 + rl;
      int k  = kc + ni * 16 + ln;
      float s = (k <= q) ? sc[i] * 0.125f : -1e30f;   // 1/sqrt(64)
      attn[((size_t)bh * SEQ + q) * SEQ + k] = s;
      atomicMaxF_lds(&rmax[rl], s);
    }
  }

  __threadfence();    // scores visible to other threads in this WG
  __syncthreads();    // rmax final

  {  // V tile 0: Vs[d][r] from d-major V: 256 x 16B, 1 per thread
    int d = tid >> 2, r8 = (tid & 3) * 8;
    copy16_g2l(&Vb[((size_t)bh * DK + d) * SEQ + r8], &Vs[0][d][r8]);
  }

  // ---- pass 2: exp + rowsum + context = E @ V ----
  v8f ca = {};
  for (int it = 0; it < nit2; ++it) {
    const int kc = it << 5, cur = it & 1;
    {  // 32x32 probabilities: 4 consecutive per thread (one float4)
      int r = tid >> 3, c4 = (tid & 7) * 4;
      size_t gi = ((size_t)bh * SEQ + q0 + r) * SEQ + kc + c4;
      float m = rmax[r];
      float4 sv = *(const float4*)&attn[gi];
      float e0 = __expf(sv.x - m), e1 = __expf(sv.y - m);
      float e2 = __expf(sv.z - m), e3 = __expf(sv.w - m);
      *(float4*)&attn[gi] = make_float4(e0, e1, e2, e3);
      uint2 p;
      p.x = (unsigned)f2bf(e0) | ((unsigned)f2bf(e1) << 16);
      p.y = (unsigned)f2bf(e2) | ((unsigned)f2bf(e3) << 16);
      *(uint2*)&Es[cur][r][c4] = p;
      atomicAdd(&rsum[r], (e0 + e1) + (e2 + e3));   // ds_add_f32
    }
    async_wait0();
    __syncthreads();
    if (it + 1 < nit2) {   // issue next V tile
      const int nk = (it + 1) << 5, nb = cur ^ 1;
      int d = tid >> 2, r8 = (tid & 3) * 8;
      copy16_g2l(&Vb[((size_t)bh * DK + d) * SEQ + nk + r8], &Vs[nb][d][r8]);
    }
    v16bf ae = frag_A(&Es[cur][mi * 16 + ln][0], half);
    v16bf bv = frag_B(&Vs[cur][ni * 16 + ln][0], half); // B[kk][d]=V[kc+kk][d]
    ca = __builtin_amdgcn_wmma_f32_16x16x32_bf16(false, ae, false, bv,
                                                 (short)0, ca, false, false);
  }

  __threadfence();    // pass-2 global writes visible within workgroup
  __syncthreads();    // rsum final

  // context write (normalized)
#pragma unroll
  for (int i = 0; i < 8; ++i) {
    int rl = mi * 16 + i + 8 * half;
    int d  = ni * 16 + ln;
    float v = ca[i] / rsum[rl];
    ctx[((size_t)(b * SEQ + q0 + rl)) * D_MODEL + h * DK + d] = v;
  }

  // ---- pass 3: normalize attn rows; zero above causal region ----
  for (int i = tid; i < 32 * (SEQ / 4); i += 256) {
    int r = i >> 8, k4 = (i & 255) * 4;
    size_t gi = ((size_t)bh * SEQ + q0 + r) * SEQ + k4;
    if (k4 < kend2) {
      float inv = 1.f / rsum[r];
      float4 v = *(const float4*)&attn[gi];
      v.x *= inv; v.y *= inv; v.z *= inv; v.w *= inv;
      *(float4*)&attn[gi] = v;
    } else {
      *(float4*)&attn[gi] = make_float4(0.f, 0.f, 0.f, 0.f);
    }
  }
}

// ---------------------------------------------------------------------------
// Kernel 3: output projection.  out = ctx @ Wo^T + bo  (fp32 out)
// ---------------------------------------------------------------------------
__global__ __launch_bounds__(256) void k_oproj(
    const float* __restrict__ ctx, const float* __restrict__ Wo,
    const float* __restrict__ bo, float* __restrict__ out)
{
  const int c0 = blockIdx.x * 64;
  const int r0 = blockIdx.y * 64;

  __shared__ alignas(16) unsigned short As[64][64];
  __shared__ alignas(16) unsigned short Bs[64][64];  // Bs[c][k]

  const int tid  = threadIdx.x;
  const int wave = tid >> 5, lane = tid & 31;
  const int half = lane >> 4, ln = lane & 15;
  const int row_t = wave >> 1;
  const int colt0 = (wave & 1) * 2;

  v8f acc0 = {}; v8f acc1 = {};

  for (int kc = 0; kc < D_MODEL; kc += 64) {
#pragma unroll
    for (int t = 0; t < 4; ++t) {
      int i = tid + t * 256;
      int r = i >> 4, c4 = (i & 15) * 4;
      float4 fa = *(const float4*)&ctx[(size_t)(r0 + r) * D_MODEL + kc + c4];
      *(uint2*)&As[r][c4] = f2bf4(fa);
      float4 fb = *(const float4*)&Wo[(size_t)(c0 + r) * D_MODEL + kc + c4];
      *(uint2*)&Bs[r][c4] = f2bf4(fb);
    }
    __syncthreads();

    v16bf a0  = frag_A(&As[row_t * 16 + ln][0], half);
    v16bf a1  = frag_A(&As[row_t * 16 + ln][32], half);
    v16bf b00 = frag_B(&Bs[colt0 * 16 + ln][0], half);
    v16bf b01 = frag_B(&Bs[colt0 * 16 + ln][32], half);
    v16bf b10 = frag_B(&Bs[(colt0 + 1) * 16 + ln][0], half);
    v16bf b11 = frag_B(&Bs[(colt0 + 1) * 16 + ln][32], half);
    acc0 = __builtin_amdgcn_wmma_f32_16x16x32_bf16(false, a0, false, b00,
                                                   (short)0, acc0, false, false);
    acc0 = __builtin_amdgcn_wmma_f32_16x16x32_bf16(false, a1, false, b01,
                                                   (short)0, acc0, false, false);
    acc1 = __builtin_amdgcn_wmma_f32_16x16x32_bf16(false, a0, false, b10,
                                                   (short)0, acc1, false, false);
    acc1 = __builtin_amdgcn_wmma_f32_16x16x32_bf16(false, a1, false, b11,
                                                   (short)0, acc1, false, false);
    __syncthreads();
  }

#pragma unroll
  for (int i = 0; i < 8; ++i) {
    int row = r0 + row_t * 16 + i + 8 * half;
    int colA = c0 + colt0 * 16 + ln;
    out[(size_t)row * D_MODEL + colA] = acc0[i] + bo[colA];
    int colB = c0 + (colt0 + 1) * 16 + ln;
    out[(size_t)row * D_MODEL + colB] = acc1[i] + bo[colB];
  }
}

// ---------------------------------------------------------------------------
extern "C" void kernel_launch(void* const* d_in, const int* in_sizes, int n_in,
                              void* d_out, int out_size, void* d_ws, size_t ws_size,
                              hipStream_t stream)
{
  (void)in_sizes; (void)n_in; (void)out_size; (void)ws_size;

  const float* x  = (const float*)d_in[0];
  const float* Wq = (const float*)d_in[1];
  const float* bq = (const float*)d_in[2];
  const float* Wk = (const float*)d_in[3];
  const float* bk = (const float*)d_in[4];
  const float* Wv = (const float*)d_in[5];
  const float* bv = (const float*)d_in[6];
  const float* Wo = (const float*)d_in[7];
  const float* bo = (const float*)d_in[8];

  // workspace carve-out: 3 x 8MB bf16 QKV + 16MB fp32 context = 40MB
  char* ws = (char*)d_ws;
  unsigned short* Qb = (unsigned short*)(ws);
  unsigned short* Kb = (unsigned short*)(ws + (size_t)8  * 1024 * 1024);
  unsigned short* Vb = (unsigned short*)(ws + (size_t)16 * 1024 * 1024);
  float*          cx = (float*)        (ws + (size_t)24 * 1024 * 1024);

  float* out  = (float*)d_out;
  float* attn = out + (size_t)BATCH * SEQ * D_MODEL;

  dim3 blk(256, 1, 1);
  k_qkv  <<<dim3(16, 64, 3),        blk, 0, stream>>>(x, Wq, bq, Wk, bk, Wv, bv,
                                                      Qb, Kb, Vb);
  k_attn <<<dim3(32, NHEAD, BATCH), blk, 0, stream>>>(Qb, Kb, Vb, attn, cx);
  k_oproj<<<dim3(16, 64, 1),        blk, 0, stream>>>(cx, Wo, bo, out);
}
